// FeedbackMetaModelBertAttention_74311524155782
// MI455X (gfx1250) — matrix operand order repaired and checked by reading, not compiled, not run on hardware
//
#include <hip/hip_runtime.h>
#include <hip/hip_bf16.h>
#include <stdint.h>

// ---------------------------------------------------------------------------
// FeedbackMetaModelBertAttention forward for MI455X (gfx1250, wave32, WMMA).
// B=32, L=512, H=512, NH=8, DH=64, NFEAT=256, NLAB=3, MAXPOS=512.
// ---------------------------------------------------------------------------

typedef __bf16 bf16;
typedef __attribute__((ext_vector_type(16))) __bf16 v16bf;
typedef __attribute__((ext_vector_type(8)))  __bf16 v8bf;
typedef __attribute__((ext_vector_type(8)))  float  v8f;

static __device__ __forceinline__ v16bf make_v16(v8bf lo, v8bf hi) {
  v16bf r;
#pragma unroll
  for (int i = 0; i < 8; ++i) { r[i] = lo[i]; r[i + 8] = hi[i]; }
  return r;
}

static __device__ __forceinline__ v8f wmma_bf16(v16bf a, v16bf b, v8f c) {
  // D = A(16x32 bf16) * B(32x16 bf16) + C(16x16 f32)
  return __builtin_amdgcn_wmma_f32_16x16x32_bf16(false, a, false, b,
                                                 (short)0, c, false, false);
}

// Async global->LDS copy (GLOBAL_LOAD_ASYNC_TO_LDS_B128, tracked by ASYNCcnt).
// GV mode: per-lane LDS byte address in VGPR (low 32 bits of shared-aperture
// generic pointer), 64-bit global address in VGPR pair, SADDR=NULL ("off").
static __device__ __forceinline__ void async_copy_b128(void* lds_ptr,
                                                       const void* gptr) {
  uint32_t lds = (uint32_t)(uintptr_t)lds_ptr;
  asm volatile("global_load_async_to_lds_b128 %0, %1, off"
               :: "v"(lds), "v"(gptr) : "memory");
}
static __device__ __forceinline__ void async_wait() {
  asm volatile("s_wait_asynccnt 0x0" ::: "memory");
}

// ---------------------------------------------------------------------------
// Weight prep: Wt[n][k] = bf16(W[k][n])
// ---------------------------------------------------------------------------
__global__ void transpose_cast_kernel(const float* __restrict__ W,
                                      bf16* __restrict__ Wt, int K, int N) {
  int idx = blockIdx.x * 256 + threadIdx.x;
  if (idx >= K * N) return;
  int kk = idx / N, nn = idx % N;
  Wt[(size_t)nn * K + kk] = (bf16)W[idx];
}

// dist_emb fp32 [1023][64] -> bf16 [1024][64], row 1023 zero padded
__global__ void dist_cast_kernel(const float* __restrict__ dist,
                                 bf16* __restrict__ out) {
  int idx = blockIdx.x * 256 + threadIdx.x; // 1024*64
  int p = idx >> 6;
  out[idx] = (p < 1023) ? (bf16)dist[idx] : (bf16)0.0f;
}

// ---------------------------------------------------------------------------
// Generic bf16 WMMA GEMM: C[M,N] = A[M,K] @ Bt[N,K]^T + bias
// A read from fp32 (Af) or bf16 (Ab) and converted while staging to LDS.
// Block tile 128x128, BK=32, 256 threads (8 waves, each 32x64 subtile).
// B tiles (and bf16 A tiles) staged with async global->LDS DMA.
// ---------------------------------------------------------------------------
__global__ __launch_bounds__(256)
void gemm_bf16_kernel(const float* __restrict__ Af, const bf16* __restrict__ Ab,
                      int lda, const bf16* __restrict__ Bt,
                      const float* __restrict__ bias,
                      float* __restrict__ outF, bf16* __restrict__ outB,
                      int M, int N, int K) {
  __shared__ bf16 Alds[128 * 40]; // pitch 40 halves = 80B (16B aligned)
  __shared__ bf16 Blds[128 * 40];

  const int t = threadIdx.x;
  const int m0 = blockIdx.y * 128;
  const int n0 = blockIdx.x * 128;
  const int wave = t >> 5, lane = t & 31, tlow = lane & 15, hi = lane >> 4;
  const int wm = (wave >> 1) * 32, wn = (wave & 1) * 64;

  v8f acc[2][4] = {};

  const int frow = t >> 1;        // 0..127
  const int fcol = (t & 1) * 16;  // 0 or 16

  for (int k0 = 0; k0 < K; k0 += 32) {
    // stage A tile (convert fp32->bf16 if needed)
    if (Af) {
      const float* src = Af + (size_t)(m0 + frow) * lda + k0 + fcol;
#pragma unroll
      for (int i = 0; i < 16; i += 4) {
        float4 f = *(const float4*)(src + i);
        bf16* d = &Alds[frow * 40 + fcol + i];
        d[0] = (bf16)f.x; d[1] = (bf16)f.y; d[2] = (bf16)f.z; d[3] = (bf16)f.w;
      }
    } else {
      const bf16* src = Ab + (size_t)(m0 + frow) * lda + k0 + fcol;
      async_copy_b128(&Alds[frow * 40 + fcol], src);
      async_copy_b128(&Alds[frow * 40 + fcol + 8], src + 8);
    }
    // stage B tile from transposed weights (contiguous, async DMA to LDS)
    {
      const bf16* src = Bt + (size_t)(n0 + frow) * K + k0 + fcol;
      async_copy_b128(&Blds[frow * 40 + fcol], src);
      async_copy_b128(&Blds[frow * 40 + fcol + 8], src + 8);
    }
    if (k0 + 32 < K) { // emits global_prefetch_b8
      if (Af) __builtin_prefetch(Af + (size_t)(m0 + frow) * lda + k0 + 32 + fcol, 0, 0);
      else    __builtin_prefetch(Ab + (size_t)(m0 + frow) * lda + k0 + 32 + fcol, 0, 0);
      __builtin_prefetch(Bt + (size_t)(n0 + frow) * K + k0 + 32 + fcol, 0, 0);
    }
    async_wait();
    __syncthreads();

    v16bf afrag[2], bfrag[4];
#pragma unroll
    for (int mi = 0; mi < 2; ++mi) {
      int row = wm + mi * 16 + tlow;
      int c0 = hi * 16; // A layout: lanes 0-15 K 0..15; lanes 16-31 K 16..31
      afrag[mi] = make_v16(*(const v8bf*)&Alds[row * 40 + c0],
                           *(const v8bf*)&Alds[row * 40 + c0 + 8]);
    }
#pragma unroll
    for (int ni = 0; ni < 4; ++ni) {
      int col = wn + ni * 16 + tlow;
      int c0 = hi * 16; // B layout: lanes 0-15 K 0..15; lanes 16-31 K 16..31
      bfrag[ni] = make_v16(*(const v8bf*)&Blds[col * 40 + c0],
                           *(const v8bf*)&Blds[col * 40 + c0 + 8]);
    }
#pragma unroll
    for (int mi = 0; mi < 2; ++mi)
#pragma unroll
      for (int ni = 0; ni < 4; ++ni)
        acc[mi][ni] = wmma_bf16(afrag[mi], bfrag[ni], acc[mi][ni]);
    __syncthreads();
  }

#pragma unroll
  for (int mi = 0; mi < 2; ++mi)
#pragma unroll
    for (int ni = 0; ni < 4; ++ni) {
      int col = n0 + wn + ni * 16 + tlow;
      float bv = bias ? bias[col] : 0.0f;
#pragma unroll
      for (int j = 0; j < 8; ++j) {
        int row = m0 + wm + mi * 16 + hi * 8 + j; // C layout row mapping
        float vv = acc[mi][ni][j] + bv;
        if (outF) outF[(size_t)row * N + col] = vv;
        if (outB) outB[(size_t)row * N + col] = (bf16)vv;
      }
    }
}

// ---------------------------------------------------------------------------
// LayerNorm over H=512 (optional residual): out = LN(in [+ res]) * g + b
// ---------------------------------------------------------------------------
__global__ __launch_bounds__(256)
void ln_kernel(const float* __restrict__ in, const float* __restrict__ res,
               const float* __restrict__ g, const float* __restrict__ bt,
               float* __restrict__ out, float eps) {
  __shared__ float red[256];
  int row = blockIdx.x, t = threadIdx.x;
  const float* p = in + (size_t)row * 512;
  float a0 = p[t], a1 = p[t + 256];
  if (res) {
    a0 += res[(size_t)row * 512 + t];
    a1 += res[(size_t)row * 512 + t + 256];
  }
  red[t] = a0 + a1;
  __syncthreads();
  for (int o = 128; o > 0; o >>= 1) { if (t < o) red[t] += red[t + o]; __syncthreads(); }
  float mu = red[0] * (1.0f / 512.0f);
  __syncthreads();
  float d0 = a0 - mu, d1 = a1 - mu;
  red[t] = d0 * d0 + d1 * d1;
  __syncthreads();
  for (int o = 128; o > 0; o >>= 1) { if (t < o) red[t] += red[t + o]; __syncthreads(); }
  float rstd = rsqrtf(red[0] * (1.0f / 512.0f) + eps);
  out[(size_t)row * 512 + t]       = d0 * rstd * g[t] + bt[t];
  out[(size_t)row * 512 + t + 256] = d1 * rstd * g[t + 256] + bt[t + 256];
}

// ---------------------------------------------------------------------------
// Classifier: logits[row][0..2] = attn[row] @ cW + cb   (NLAB=3)
// ---------------------------------------------------------------------------
__global__ __launch_bounds__(256)
void cls_kernel(const float* __restrict__ attn, const float* __restrict__ cW,
                const float* __restrict__ cb, float* __restrict__ logits) {
  __shared__ float red[3][256];
  int row = blockIdx.x, t = threadIdx.x;
  const float* a = attn + (size_t)row * 512;
  float p0 = 0.f, p1 = 0.f, p2 = 0.f;
  for (int i = t; i < 512; i += 256) {
    float av = a[i];
    p0 += av * cW[i * 3 + 0]; p1 += av * cW[i * 3 + 1]; p2 += av * cW[i * 3 + 2];
  }
  red[0][t] = p0; red[1][t] = p1; red[2][t] = p2;
  __syncthreads();
  for (int o = 128; o > 0; o >>= 1) {
    if (t < o) { red[0][t] += red[0][t + o]; red[1][t] += red[1][t + o]; red[2][t] += red[2][t + o]; }
    __syncthreads();
  }
  if (t == 0) {
    logits[row * 3 + 0] = red[0][0] + cb[0];
    logits[row * 3 + 1] = red[1][0] + cb[1];
    logits[row * 3 + 2] = red[2][0] + cb[2];
  }
}

// ---------------------------------------------------------------------------
// Cross entropy (ignore_index=-1): accum[0]+=sum(nll), accum[1]+=count
// ---------------------------------------------------------------------------
__global__ __launch_bounds__(256)
void loss_kernel(const float* __restrict__ logits, const int* __restrict__ labels,
                 float* __restrict__ accum) {
  __shared__ float rs[256], rc[256];
  int i = blockIdx.x * 256 + threadIdx.x; // 16384 tokens
  float l0 = logits[i * 3], l1 = logits[i * 3 + 1], l2 = logits[i * 3 + 2];
  float mx = fmaxf(l0, fmaxf(l1, l2));
  float lse = mx + logf(__expf(l0 - mx) + __expf(l1 - mx) + __expf(l2 - mx));
  int lab = labels[i];
  float nll = 0.0f, cnt = 0.0f;
  if (lab >= 0) {
    float sel = (lab == 0) ? l0 : ((lab == 1) ? l1 : l2);
    nll = lse - sel; cnt = 1.0f;
  }
  rs[threadIdx.x] = nll; rc[threadIdx.x] = cnt;
  __syncthreads();
  for (int o = 128; o > 0; o >>= 1) {
    if (threadIdx.x < o) { rs[threadIdx.x] += rs[threadIdx.x + o]; rc[threadIdx.x] += rc[threadIdx.x + o]; }
    __syncthreads();
  }
  if (threadIdx.x == 0) { atomicAdd(accum, rs[0]); atomicAdd(accum + 1, rc[0]); }
}

__global__ void finalize_loss_kernel(const float* __restrict__ accum,
                                     float* __restrict__ out) {
  out[0] = accum[0] / fmaxf(accum[1], 1.0f);
}

// ---------------------------------------------------------------------------
// Flash attention with relative_key bias, one workgroup per (b,h).
// scores = (q@k^T + q@dist_emb[l-r+511]^T)/8 + mask_bias; ctx = softmax @ v
// ---------------------------------------------------------------------------
__global__ __launch_bounds__(256)
void attn_kernel(const bf16* __restrict__ q, const bf16* __restrict__ k,
                 const bf16* __restrict__ v, const bf16* __restrict__ dist,
                 const float* __restrict__ mask, bf16* __restrict__ ctx) {
  __shared__ bf16 Klds[128 * 72];       // [r][d]  pitch 72 halves (144B)
  __shared__ bf16 Vlds[64 * 136];       // [d][r]  pitch 136 halves (272B)
  __shared__ bf16 qdlds[8][16 * 48];    // per-wave bias tile [li][c]
  __shared__ bf16 plds[8][16 * 40];     // per-wave P staging [li][rj]

  const int b = blockIdx.x >> 3, h = blockIdx.x & 7;
  const int t = threadIdx.x, wave = t >> 5, lane = t & 31;
  const int tlow = lane & 15, hi = lane >> 4;

  for (int qt = 0; qt < 4; ++qt) {
    const int l0 = (qt * 8 + wave) * 16;
    // Q fragments (A layout) straight from global
    const bf16* qrow = q + ((size_t)(b * 512 + l0 + tlow) * 512) + h * 64;
    v16bf a0, a1;
    {
      int c0 = hi * 16;
      a0 = make_v16(*(const v8bf*)(qrow + c0),      *(const v8bf*)(qrow + c0 + 8));
      a1 = make_v16(*(const v8bf*)(qrow + c0 + 32), *(const v8bf*)(qrow + c0 + 40));
    }
    float m_run[8], l_run[8];
    v8f cacc[4] = {};
#pragma unroll
    for (int j = 0; j < 8; ++j) { m_run[j] = -1e30f; l_run[j] = 0.0f; }

    for (int kc = 0; kc < 512; kc += 128) {
      __syncthreads();
      { // cooperative stage: K row-major (async DMA), V transposed (VALU)
        int rr = t >> 1;          // 0..127
        int dh = (t & 1) * 32;    // 0 or 32
        const bf16* ks = k + ((size_t)(b * 512 + kc + rr) * 512) + h * 64 + dh;
#pragma unroll
        for (int i = 0; i < 32; i += 8)
          async_copy_b128(&Klds[rr * 72 + dh + i], ks + i);
        const bf16* vs = v + ((size_t)(b * 512 + kc + rr) * 512) + h * 64 + dh;
#pragma unroll
        for (int i = 0; i < 32; ++i)
          Vlds[(dh + i) * 136 + rr] = vs[i];
      }
      async_wait();
      __syncthreads();

      for (int kb = 0; kb < 128; kb += 32) {
        const int r0 = kc + kb;
        // ---- scores S[16,32] = Q @ K^T (4 WMMAs) ----
        v8f s0 = {}, s1 = {};
        {
          int c0 = hi * 16;
          int col = kb + tlow;
          v16bf b0 = make_v16(*(const v8bf*)&Klds[col * 72 + c0],
                              *(const v8bf*)&Klds[col * 72 + c0 + 8]);
          v16bf b1 = make_v16(*(const v8bf*)&Klds[col * 72 + c0 + 32],
                              *(const v8bf*)&Klds[col * 72 + c0 + 40]);
          s0 = wmma_bf16(a0, b0, s0);
          s0 = wmma_bf16(a1, b1, s0);
          int col2 = kb + 16 + tlow;
          v16bf b2 = make_v16(*(const v8bf*)&Klds[col2 * 72 + c0],
                              *(const v8bf*)&Klds[col2 * 72 + c0 + 8]);
          v16bf b3 = make_v16(*(const v8bf*)&Klds[col2 * 72 + c0 + 32],
                              *(const v8bf*)&Klds[col2 * 72 + c0 + 40]);
          s1 = wmma_bf16(a0, b2, s1);
          s1 = wmma_bf16(a1, b3, s1);
        }
        // ---- relative bias qd[16,48] = Q @ dist[p]^T (6 WMMAs) ----
        {
          int p_base = l0 - r0 + 480; // p = l - r + 511, c = li - rj + 31
#pragma unroll
          for (int t3 = 0; t3 < 3; ++t3) {
            int p = p_base + t3 * 16 + tlow;
            const bf16* dr = dist + (size_t)p * 64;
            int c0 = hi * 16;
            v16bf d0 = make_v16(*(const v8bf*)(dr + c0),      *(const v8bf*)(dr + c0 + 8));
            v16bf d1 = make_v16(*(const v8bf*)(dr + c0 + 32), *(const v8bf*)(dr + c0 + 40));
            v8f qd = {};
            qd = wmma_bf16(a0, d0, qd);
            qd = wmma_bf16(a1, d1, qd);
#pragma unroll
            for (int j = 0; j < 8; ++j) {
              int li = j + hi * 8;
              qdlds[wave][li * 48 + t3 * 16 + tlow] = (bf16)(qd[j] * 0.125f);
            }
          }
        }
        // ---- scale + bias + mask ----
        {
          float mb0 = (1.0f - mask[b * 512 + r0 + tlow]) * -10000.0f;
          float mb1 = (1.0f - mask[b * 512 + r0 + 16 + tlow]) * -10000.0f;
#pragma unroll
          for (int j = 0; j < 8; ++j) {
            int li = j + hi * 8;
            s0[j] = s0[j] * 0.125f + (float)qdlds[wave][li * 48 + (li - tlow + 31)] + mb0;
            s1[j] = s1[j] * 0.125f + (float)qdlds[wave][li * 48 + (li - 16 - tlow + 31)] + mb1;
          }
        }
        // ---- online softmax (row reduce over 16-lane halves) ----
        float p0v[8], p1v[8], alpha[8];
#pragma unroll
        for (int j = 0; j < 8; ++j) {
          float rm = fmaxf(s0[j], s1[j]);
#pragma unroll
          for (int m = 1; m < 16; m <<= 1) rm = fmaxf(rm, __shfl_xor(rm, m, 32));
          float mn = fmaxf(m_run[j], rm);
          alpha[j] = __expf(m_run[j] - mn);
          float e0 = __expf(s0[j] - mn), e1 = __expf(s1[j] - mn);
          float rsum = e0 + e1;
#pragma unroll
          for (int m = 1; m < 16; m <<= 1) rsum += __shfl_xor(rsum, m, 32);
          l_run[j] = l_run[j] * alpha[j] + rsum;
          m_run[j] = mn;
          p0v[j] = e0; p1v[j] = e1;
        }
#pragma unroll
        for (int dn = 0; dn < 4; ++dn)
#pragma unroll
          for (int j = 0; j < 8; ++j) cacc[dn][j] *= alpha[j];
        // ---- P (C layout) -> LDS -> A layout ----
#pragma unroll
        for (int j = 0; j < 8; ++j) {
          int li = j + hi * 8;
          plds[wave][li * 40 + tlow]      = (bf16)p0v[j];
          plds[wave][li * 40 + 16 + tlow] = (bf16)p1v[j];
        }
        v16bf pa;
        {
          int c0 = hi * 16;
          pa = make_v16(*(const v8bf*)&plds[wave][tlow * 40 + c0],
                        *(const v8bf*)&plds[wave][tlow * 40 + c0 + 8]);
        }
        // ---- ctx += P @ V (4 WMMAs) ----
#pragma unroll
        for (int dn = 0; dn < 4; ++dn) {
          int dcol = dn * 16 + tlow;
          v16bf vb = make_v16(*(const v8bf*)&Vlds[dcol * 136 + kb + hi * 16],
                              *(const v8bf*)&Vlds[dcol * 136 + kb + hi * 16 + 8]);
          cacc[dn] = wmma_bf16(pa, vb, cacc[dn]);
        }
      }
    }
    // ---- normalize + write ctx (bf16, row-major [B*L, H]) ----
#pragma unroll
    for (int j = 0; j < 8; ++j) {
      float inv = 1.0f / l_run[j];
      int l = l0 + j + hi * 8;
      bf16* orow = ctx + ((size_t)(b * 512 + l) * 512) + h * 64;
#pragma unroll
      for (int dn = 0; dn < 4; ++dn)
        orow[dn * 16 + tlow] = (bf16)(cacc[dn][j] * inv);
    }
  }
}

// ---------------------------------------------------------------------------
extern "C" void kernel_launch(void* const* d_in, const int* in_sizes, int n_in,
                              void* d_out, int out_size, void* d_ws, size_t ws_size,
                              hipStream_t stream) {
  (void)in_sizes; (void)n_in; (void)out_size; (void)ws_size;

  const float* meta = (const float*)d_in[0];
  const float* span = (const float*)d_in[2];
  const int*   labels = (const int*)d_in[4];
  const float* pW = (const float*)d_in[5];  const float* pb = (const float*)d_in[6];
  const float* g1 = (const float*)d_in[7];  const float* b1 = (const float*)d_in[8];
  const float* Wq = (const float*)d_in[9];  const float* bq = (const float*)d_in[10];
  const float* Wk = (const float*)d_in[11]; const float* bk = (const float*)d_in[12];
  const float* Wv = (const float*)d_in[13]; const float* bv = (const float*)d_in[14];
  const float* dist = (const float*)d_in[15];
  const float* Wo = (const float*)d_in[16]; const float* bo = (const float*)d_in[17];
  const float* g2 = (const float*)d_in[18]; const float* b2 = (const float*)d_in[19];
  const float* cW = (const float*)d_in[20]; const float* cb = (const float*)d_in[21];
  float* out = (float*)d_out; // 49152 logits + 1 loss

  const int M = 32 * 512; // 16384 tokens
  char* ws = (char*)d_ws;
  size_t off = 0;
  auto alloc = [&](size_t bytes) -> void* {
    void* p = ws + off;
    off = (off + bytes + 255) & ~(size_t)255;
    return p;
  };
  float* y    = (float*)alloc((size_t)M * 512 * 4); // proj pre-LN / attn_out reuse
  float* x    = (float*)alloc((size_t)M * 512 * 4); // LN1 output (residual)
  float* o    = (float*)alloc((size_t)M * 512 * 4); // ctx @ Wo + bo
  bf16* qb    = (bf16*)alloc((size_t)M * 512 * 2);
  bf16* kb    = (bf16*)alloc((size_t)M * 512 * 2);
  bf16* vb    = (bf16*)alloc((size_t)M * 512 * 2);
  bf16* ctxb  = (bf16*)alloc((size_t)M * 512 * 2);
  bf16* pWt   = (bf16*)alloc((size_t)512 * 256 * 2);
  bf16* Wqt   = (bf16*)alloc((size_t)512 * 512 * 2);
  bf16* Wkt   = (bf16*)alloc((size_t)512 * 512 * 2);
  bf16* Wvt   = (bf16*)alloc((size_t)512 * 512 * 2);
  bf16* Wot   = (bf16*)alloc((size_t)512 * 512 * 2);
  bf16* distb = (bf16*)alloc((size_t)1024 * 64 * 2);
  float* accum = (float*)alloc(256);

  // weight prep (cast + transpose to [N][K])
  transpose_cast_kernel<<<(256 * 512 + 255) / 256, 256, 0, stream>>>(pW, pWt, 256, 512);
  transpose_cast_kernel<<<(512 * 512 + 255) / 256, 256, 0, stream>>>(Wq, Wqt, 512, 512);
  transpose_cast_kernel<<<(512 * 512 + 255) / 256, 256, 0, stream>>>(Wk, Wkt, 512, 512);
  transpose_cast_kernel<<<(512 * 512 + 255) / 256, 256, 0, stream>>>(Wv, Wvt, 512, 512);
  transpose_cast_kernel<<<(512 * 512 + 255) / 256, 256, 0, stream>>>(Wo, Wot, 512, 512);
  dist_cast_kernel<<<(1024 * 64) / 256, 256, 0, stream>>>(dist, distb);

  dim3 gg(512 / 128, M / 128); // (4, 128)
  // y = meta @ pW + pb ; x = LN1(y)
  gemm_bf16_kernel<<<gg, 256, 0, stream>>>(meta, nullptr, 256, pWt, pb, y, nullptr, M, 512, 256);
  ln_kernel<<<M, 256, 0, stream>>>(y, nullptr, g1, b1, x, 1e-7f);
  // q/k/v = x @ W + b (bf16 outputs)
  gemm_bf16_kernel<<<gg, 256, 0, stream>>>(x, nullptr, 512, Wqt, bq, nullptr, qb, M, 512, 512);
  gemm_bf16_kernel<<<gg, 256, 0, stream>>>(x, nullptr, 512, Wkt, bk, nullptr, kb, M, 512, 512);
  gemm_bf16_kernel<<<gg, 256, 0, stream>>>(x, nullptr, 512, Wvt, bv, nullptr, vb, M, 512, 512);
  // attention with relative_key bias
  attn_kernel<<<256, 256, 0, stream>>>(qb, kb, vb, distb, span, ctxb);
  // o = ctx @ Wo + bo ; attn_out = LN2(o + x) (into y) ; logits
  gemm_bf16_kernel<<<gg, 256, 0, stream>>>(nullptr, ctxb, 512, Wot, bo, o, nullptr, M, 512, 512);
  ln_kernel<<<M, 256, 0, stream>>>(o, x, g2, b2, y, 1e-12f);
  cls_kernel<<<M, 256, 0, stream>>>(y, cW, cb, out);
  // loss
  hipMemsetAsync(accum, 0, 8, stream);
  loss_kernel<<<M / 256, 256, 0, stream>>>(out, labels, accum);
  finalize_loss_kernel<<<1, 1, 0, stream>>>(accum, out + (size_t)M * 3);
}
